// Attention_85693187490031
// MI455X (gfx1250) — matrix operand order repaired and checked by reading, not compiled, not run on hardware
//
#include <hip/hip_runtime.h>
#include <hip/hip_bf16.h>
#include <math.h>

#define B_   32
#define N_   576
#define C_   768
#define H_   12
#define HD_  64
#define SCALE_ 0.125f   // 1/sqrt(64)

typedef __attribute__((ext_vector_type(16))) __bf16 v16bf;
typedef __attribute__((ext_vector_type(8)))  float  v8f;

static __device__ inline __bf16 to_bf(float v)  { return (__bf16)v; }
static __device__ inline __bf16 to_bf(__bf16 v) { return v; }

static __device__ inline v8f vzero() {
  v8f z;
#pragma unroll
  for (int i = 0; i < 8; ++i) z[i] = 0.0f;
  return z;
}

static __device__ inline v8f wmma_bf16(v16bf a, v16bf b, v8f c) {
  return __builtin_amdgcn_wmma_f32_16x16x32_bf16(false, a, false, b, (short)0, c,
                                                 false, false);
}

// ---- fragment loaders (contiguous K order k = 16*half + s, wave32) -------------------

// A-matrix 16x32 (MxK), 16-bit: lane half selects K-block, m = lane&15.
// K runs contiguously in memory per lane -> 2x b128 when T==__bf16.
template <typename T>
static __device__ inline v16bf load_a_frag(const T* A, int lda, int m0, int k0,
                                           int lane) {
  const int hl = (lane >> 4) & 1;
  const int m  = lane & 15;
  const T* row = A + (size_t)(m0 + m) * lda + k0;
  v16bf a;
#pragma unroll
  for (int p = 0; p < 8; ++p) {
    int kb = 2 * p + 16 * hl;
    a[2 * p]     = to_bf(row[kb]);
    a[2 * p + 1] = to_bf(row[kb + 1]);
  }
  return a;
}

// B-matrix 32x16 where B[k][n] = W[n][k] (W stored row-major N x K) —
// the X @ W^T pattern; K runs contiguously in memory per lane.
template <typename T>
static __device__ inline v16bf load_b_frag_t(const T* W, int ldw, int k0,
                                             int n0, int lane) {
  const int g = (lane >> 4) & 1;
  const int n = lane & 15;
  const T* row = W + (size_t)(n0 + n) * ldw + k0;
  v16bf b;
#pragma unroll
  for (int v = 0; v < 8; ++v) {
    int k = 2 * v + 16 * g;
    b[2 * v]     = to_bf(row[k]);
    b[2 * v + 1] = to_bf(row[k + 1]);
  }
  return b;
}

__device__ inline int blk_of(int p) {  // block index on 24x24 grid, 4x4 blocks
  return (p / 96) * 6 + ((p % 24) >> 2);
}

// ---- kernel 0: fp32 -> bf16 pre-conversion --------------------------------
__global__ __launch_bounds__(256) void cvt_kernel(const float* __restrict__ s,
                                                  __bf16* __restrict__ d,
                                                  int n) {
  int i = (blockIdx.x * 256 + threadIdx.x) * 4;
  if (i + 3 < n) {
#pragma unroll
    for (int j = 0; j < 4; ++j) d[i + j] = (__bf16)s[i + j];
  }
}

// ---- kernel 1: qkv = x @ qkv_w^T + b --------------------------------------
// 32x64 output tile per wave; Q,K stored (B,H,N,64); V stored (B,H,64,N).
__global__ __launch_bounds__(32) void qkv_kernel(
    const __bf16* __restrict__ x, const __bf16* __restrict__ w,
    const float* __restrict__ bias, __bf16* __restrict__ qb,
    __bf16* __restrict__ kb, __bf16* __restrict__ vt) {
  const int lane = threadIdx.x & 31;
  const int m0   = blockIdx.x * 32;   // row in [0, B*N)
  const int nt   = blockIdx.y;        // 64-col tile in [0, 36)
  const int j0   = nt * 64;

  v8f acc[2][4];
#pragma unroll
  for (int mi = 0; mi < 2; ++mi)
#pragma unroll
    for (int t = 0; t < 4; ++t) acc[mi][t] = vzero();

  for (int kk = 0; kk < C_; kk += 32) {
    v16bf a0 = load_a_frag(x, C_, m0,      kk, lane);
    v16bf a1 = load_a_frag(x, C_, m0 + 16, kk, lane);
#pragma unroll
    for (int t = 0; t < 4; ++t) {
      v16bf b = load_b_frag_t(w, C_, kk, j0 + 16 * t, lane);
      acc[0][t] = wmma_bf16(a0, b, acc[0][t]);
      acc[1][t] = wmma_bf16(a1, b, acc[1][t]);
    }
  }

  const int s = nt / 12;      // 0=q 1=k 2=v
  const int h = nt % 12;
  const int hl = (lane >> 4) & 1, lm = lane & 15;
  // 32-aligned row tile never crosses a batch boundary (N_ % 32 == 0)
  const int bidx = m0 / N_;
#pragma unroll
  for (int mi = 0; mi < 2; ++mi)
#pragma unroll
    for (int t = 0; t < 4; ++t) {
      float bv = bias[j0 + 16 * t + lm];
      int d = 16 * t + lm;
#pragma unroll
      for (int r = 0; r < 8; ++r) {
        int i = m0 + mi * 16 + r + 8 * hl;
        int n = i - bidx * N_;
        __bf16 val = (__bf16)(acc[mi][t][r] + bv);
        if (s == 2) {  // V transposed: (b,h,d,n)
          vt[(((size_t)bidx * H_ + h) * HD_ + d) * N_ + n] = val;
        } else {
          __bf16* dst = (s == 0) ? qb : kb;
          dst[(((size_t)bidx * H_ + h) * N_ + n) * HD_ + d] = val;
        }
      }
    }
}

// ---- kernel 2: flash attention with block-causal mask, one wave / q-tile ---
__global__ __launch_bounds__(32) void attn_kernel(
    const __bf16* __restrict__ qb, const __bf16* __restrict__ kb,
    const __bf16* __restrict__ vt, __bf16* __restrict__ ab) {
  __shared__ __bf16 pSh[16 * 32];
  const int lane = threadIdx.x & 31;
  const int bh   = blockIdx.x;            // [0, B*H)
  const int qt   = blockIdx.y;            // [0, N/16)
  const size_t base = (size_t)bh * N_ * HD_;
  const int hl = (lane >> 4) & 1, lm = lane & 15;

  const __bf16* Q = qb + base;   // (n, d) row-major
  const __bf16* K = kb + base;   // (n, d) row-major
  const __bf16* V = vt + base;   // (d, n) row-major (transposed)

  v16bf aq0 = load_a_frag(Q, HD_, qt * 16, 0,  lane);
  v16bf aq1 = load_a_frag(Q, HD_, qt * 16, 32, lane);

  int biq[8];
#pragma unroll
  for (int r = 0; r < 8; ++r) biq[r] = blk_of(qt * 16 + r + 8 * hl);

  float mrun[8], lrun[8];
  v8f o[4];
#pragma unroll
  for (int r = 0; r < 8; ++r) { mrun[r] = -INFINITY; lrun[r] = 0.0f; }
#pragma unroll
  for (int t = 0; t < 4; ++t) o[t] = vzero();

  for (int kt = 0; kt < N_; kt += 32) {
    v8f s0 = vzero(), s1 = vzero();
    {
      v16bf b00 = load_b_frag_t(K, HD_, 0,  kt,      lane);
      v16bf b01 = load_b_frag_t(K, HD_, 32, kt,      lane);
      v16bf b10 = load_b_frag_t(K, HD_, 0,  kt + 16, lane);
      v16bf b11 = load_b_frag_t(K, HD_, 32, kt + 16, lane);
      s0 = wmma_bf16(aq0, b00, s0);
      s0 = wmma_bf16(aq1, b01, s0);
      s1 = wmma_bf16(aq0, b10, s1);
      s1 = wmma_bf16(aq1, b11, s1);
    }

    const int bik0 = blk_of(kt + lm);
    const int bik1 = blk_of(kt + 16 + lm);
    float mloc[8];
#pragma unroll
    for (int r = 0; r < 8; ++r) {
      float e0 = (bik0 > biq[r]) ? -INFINITY : s0[r] * SCALE_;
      float e1 = (bik1 > biq[r]) ? -INFINITY : s1[r] * SCALE_;
      s0[r] = e0; s1[r] = e1;
      mloc[r] = fmaxf(e0, e1);
    }
#pragma unroll
    for (int off = 1; off <= 8; off <<= 1)
#pragma unroll
      for (int r = 0; r < 8; ++r)
        mloc[r] = fmaxf(mloc[r], __shfl_xor(mloc[r], off, 32));

    float alpha[8], msafe[8];
#pragma unroll
    for (int r = 0; r < 8; ++r) {
      float mnew = fmaxf(mrun[r], mloc[r]);
      float ms   = (mnew == -INFINITY) ? 0.0f : mnew;
      alpha[r] = __expf(mrun[r] - ms);  // exp(-inf)=0 on first contribution
      mrun[r]  = mnew;
      msafe[r] = ms;
    }

    float psum[8];
#pragma unroll
    for (int r = 0; r < 8; ++r) {
      float p0 = __expf(s0[r] - msafe[r]);
      float p1 = __expf(s1[r] - msafe[r]);
      s0[r] = p0; s1[r] = p1;
      psum[r] = p0 + p1;
    }
#pragma unroll
    for (int off = 1; off <= 8; off <<= 1)
#pragma unroll
      for (int r = 0; r < 8; ++r)
        psum[r] += __shfl_xor(psum[r], off, 32);

#pragma unroll
    for (int r = 0; r < 8; ++r) lrun[r] = lrun[r] * alpha[r] + psum[r];
#pragma unroll
    for (int t = 0; t < 4; ++t)
#pragma unroll
      for (int r = 0; r < 8; ++r) o[t][r] *= alpha[r];

    // C-layout -> A-layout reshape of P via LDS (single wave: DS ops ordered)
#pragma unroll
    for (int r = 0; r < 8; ++r) {
      pSh[(r + 8 * hl) * 32 + lm]      = (__bf16)s0[r];
      pSh[(r + 8 * hl) * 32 + 16 + lm] = (__bf16)s1[r];
    }
    __syncthreads();
    v16bf ap = load_a_frag((const __bf16*)pSh, 32, 0, 0, lane);
#pragma unroll
    for (int t = 0; t < 4; ++t) {
      // B[k][n] = V^T[d = t*16+n][key = kt+k]  -> contiguous-K pattern
      v16bf bv = load_b_frag_t(V, N_, kt, t * 16, lane);
      o[t] = wmma_bf16(ap, bv, o[t]);
    }
    __syncthreads();
  }

  const int b = bh / H_, h = bh % H_;
  float inv[8];
#pragma unroll
  for (int r = 0; r < 8; ++r) inv[r] = (lrun[r] > 0.0f) ? 1.0f / lrun[r] : 0.0f;
#pragma unroll
  for (int t = 0; t < 4; ++t)
#pragma unroll
    for (int r = 0; r < 8; ++r) {
      int n = qt * 16 + r + 8 * hl;
      ab[((size_t)b * N_ + n) * C_ + h * HD_ + t * 16 + lm] =
          (__bf16)(o[t][r] * inv[r]);
    }
}

// ---- kernel 3: out = attn @ proj_w^T + proj_b (fp32 result) ---------------
__global__ __launch_bounds__(32) void proj_kernel(
    const __bf16* __restrict__ a, const __bf16* __restrict__ w,
    const float* __restrict__ bias, float* __restrict__ out) {
  const int lane = threadIdx.x & 31;
  const int m0   = blockIdx.x * 32;
  const int j0   = blockIdx.y * 64;

  v8f acc[2][4];
#pragma unroll
  for (int mi = 0; mi < 2; ++mi)
#pragma unroll
    for (int t = 0; t < 4; ++t) acc[mi][t] = vzero();

  for (int kk = 0; kk < C_; kk += 32) {
    v16bf a0 = load_a_frag(a, C_, m0,      kk, lane);
    v16bf a1 = load_a_frag(a, C_, m0 + 16, kk, lane);
#pragma unroll
    for (int t = 0; t < 4; ++t) {
      v16bf b = load_b_frag_t(w, C_, kk, j0 + 16 * t, lane);
      acc[0][t] = wmma_bf16(a0, b, acc[0][t]);
      acc[1][t] = wmma_bf16(a1, b, acc[1][t]);
    }
  }

  const int hl = (lane >> 4) & 1, lm = lane & 15;
#pragma unroll
  for (int mi = 0; mi < 2; ++mi)
#pragma unroll
    for (int t = 0; t < 4; ++t) {
      float bv = bias[j0 + 16 * t + lm];
#pragma unroll
      for (int r = 0; r < 8; ++r)
        out[(size_t)(m0 + mi * 16 + r + 8 * hl) * C_ + j0 + 16 * t + lm] =
            acc[mi][t][r] + bv;
    }
}

extern "C" void kernel_launch(void* const* d_in, const int* in_sizes, int n_in,
                              void* d_out, int out_size, void* d_ws,
                              size_t ws_size, hipStream_t stream) {
  const float* x      = (const float*)d_in[0];
  const float* qkv_w  = (const float*)d_in[1];
  const float* qkv_b  = (const float*)d_in[2];
  const float* proj_w = (const float*)d_in[3];
  const float* proj_b = (const float*)d_in[4];

  const size_t SZ  = (size_t)B_ * N_ * C_;     // 14,155,776 (= B*H*N*64)
  const size_t WQ  = (size_t)3 * C_ * C_;      // 1,769,472
  const size_t WP  = (size_t)C_ * C_;          // 589,824

  __bf16* qb  = (__bf16*)d_ws;
  __bf16* kb  = qb  + SZ;
  __bf16* vt  = kb  + SZ;   // V stored transposed (B,H,64,N)
  __bf16* ab  = vt  + SZ;   // attention output (B,N,C)
  __bf16* xb  = ab  + SZ;   // x in bf16
  __bf16* wqb = xb  + SZ;   // qkv_w in bf16
  __bf16* wpb = wqb + WQ;   // proj_w in bf16

  cvt_kernel<<<dim3((int)(SZ / 1024)), dim3(256), 0, stream>>>(x, xb, (int)SZ);
  cvt_kernel<<<dim3((int)(WQ / 1024)), dim3(256), 0, stream>>>(qkv_w, wqb, (int)WQ);
  cvt_kernel<<<dim3((int)(WP / 1024)), dim3(256), 0, stream>>>(proj_w, wpb, (int)WP);

  qkv_kernel<<<dim3((B_ * N_) / 32, (3 * C_) / 64), dim3(32), 0, stream>>>(
      xb, wqb, qkv_b, qb, kb, vt);
  attn_kernel<<<dim3(B_ * H_, N_ / 16), dim3(32), 0, stream>>>(qb, kb, vt, ab);
  proj_kernel<<<dim3((B_ * N_) / 32, C_ / 64), dim3(32), 0, stream>>>(
      ab, wpb, proj_b, (float*)d_out);
}
